// SAGPooling_Hierarchical_76089640616131
// MI455X (gfx1250) — matrix operand order repaired and checked by reading, not compiled, not run on hardware
//
#include <hip/hip_runtime.h>

namespace {

constexpr int kB   = 64;
constexpr int kN   = 512;
constexpr int kDeg = 16;
constexpr int kE   = kB * kN * kDeg;   // 524288 edges
constexpr int kHid = 128;

typedef __attribute__((ext_vector_type(16))) _Float16 v16h;
typedef __attribute__((ext_vector_type(8)))  _Float16 v8h;
typedef __attribute__((ext_vector_type(8)))  float    v8f;

// ---------------------------------------------------------------- utilities
__global__ void k_zero_f32(float* p, long long count) {
  long long i = (long long)blockIdx.x * blockDim.x + threadIdx.x;
  if (i < count) p[i] = 0.0f;
}

__global__ void k_init_edges(const int* __restrict__ src0, const int* __restrict__ dst0,
                             int* __restrict__ src, int* __restrict__ dst,
                             float* __restrict__ emask) {
  int e = blockIdx.x * blockDim.x + threadIdx.x;
  if (e >= kE) return;
  src[e] = src0[e];
  dst[e] = dst0[e];
  emask[e] = 1.0f;
}

// ---------------------------------------------------------------- GCN pieces
__global__ void k_deg_accum(const int* __restrict__ dst, const float* __restrict__ emask,
                            float* __restrict__ deg) {
  int e = blockIdx.x * blockDim.x + threadIdx.x;
  if (e >= kE) return;
  float m = emask[e];
  if (m != 0.0f) atomicAdd(&deg[dst[e]], m);
}

__global__ void k_dinv(const float* __restrict__ deg, float* __restrict__ dinv, int n) {
  int i = blockIdx.x * blockDim.x + threadIdx.x;
  if (i < n) dinv[i] = rsqrtf(deg[i] + 1.0f);   // +1 self loop
}

// ---- GEMM staging: A[M,K] f32 -> Ah[M,Kp] f16 (zero-padded K) ----
__global__ void k_pack_a(const float* __restrict__ A, _Float16* __restrict__ Ah,
                         long long total, int K, int Kp) {
  long long i = (long long)blockIdx.x * blockDim.x + threadIdx.x;
  if (i >= total) return;
  int k = (int)(i % Kp);
  long long row = i / Kp;
  Ah[i] = (k < K) ? (_Float16)A[row * K + k] : (_Float16)0.0f;
}

// ---- W[K,N] f32 -> Wt[N,Kp] f16 (transposed, zero-padded K) ----
__global__ void k_pack_wt(const float* __restrict__ W, _Float16* __restrict__ Wt,
                          long long total, int K, int N, int Kp) {
  long long i = (long long)blockIdx.x * blockDim.x + threadIdx.x;
  if (i >= total) return;
  int k = (int)(i % Kp);
  int n = (int)(i / Kp);
  Wt[i] = (k < K) ? (_Float16)W[(size_t)k * N + n] : (_Float16)0.0f;
}

// C[M,N] = Ah[M,KP] * Wt[N,KP]^T via f16 WMMA, f32 accumulate. KP % 32 == 0.
// grid = (M/16, ceil(N/64)), block = 128 (4 waves; one 16x16 tile per wave).
// Fragment loads are contiguous 16B vectors (no divergence, EXEC stays all-ones).
template <int KP>
__global__ void k_gemm_wmma_h(const _Float16* __restrict__ Ah,
                              const _Float16* __restrict__ Wt,
                              float* __restrict__ C, int N) {
  int wid  = threadIdx.x >> 5;
  int lane = threadIdx.x & 31;
  int tm = blockIdx.x;
  int tn = blockIdx.y * 4 + wid;
  if (tn * 16 >= N) return;          // wave-uniform exit
  int m = lane & 15;                 // row (A) / col (B,C)
  int g = lane >> 4;                 // lane-group
  const _Float16* rowA = Ah + (size_t)(tm * 16 + m) * KP;
  const _Float16* rowB = Wt + (size_t)(tn * 16 + m) * KP;
  v8f acc = {};
#pragma unroll
  for (int kk = 0; kk < KP; kk += 32) {
    // A 16x32: lane g=0 -> K {8g..8g+7, 16+8g..16+8g+7}; both runs contiguous
    v8h a0 = *(const v8h*)(rowA + kk + 8 * g);
    v8h a1 = *(const v8h*)(rowA + kk + 16 + 8 * g);
    // B 32x16: lane holds K {8g..8g+7, 16+8g..16+8g+7} of one column (Wt row), matching A
    v8h b0 = *(const v8h*)(rowB + kk + 8 * g);
    v8h b1 = *(const v8h*)(rowB + kk + 16 + 8 * g);
    v16h a = __builtin_shufflevector(a0, a1, 0, 1, 2, 3, 4, 5, 6, 7,
                                     8, 9, 10, 11, 12, 13, 14, 15);
    v16h b = __builtin_shufflevector(b0, b1, 0, 1, 2, 3, 4, 5, 6, 7,
                                     8, 9, 10, 11, 12, 13, 14, 15);
    acc = __builtin_amdgcn_wmma_f32_16x16x32_f16(false, a, false, b, (short)0, acc,
                                                 false, false);
  }
  // C/D layout: VGPR r -> row r + 8*g, col = lane%16
#pragma unroll
  for (int r = 0; r < 8; ++r) {
    C[(size_t)(tm * 16 + r + 8 * g) * N + tn * 16 + m] = acc[r];
  }
}

// agg[dst,f] += hw[src,f] * dinv[src]*dinv[dst]*emask  (one thread per edge-feature)
__global__ void k_edge_feat_agg(const int* __restrict__ src, const int* __restrict__ dst,
                                const float* __restrict__ emask, const float* __restrict__ dinv,
                                const float* __restrict__ hw, float* __restrict__ agg,
                                long long total) {
  long long i = (long long)blockIdx.x * blockDim.x + threadIdx.x;
  if (i >= total) return;
  int f = (int)(i & (kHid - 1));
  int e = (int)(i >> 7);
  float msk = emask[e];
  if (msk == 0.0f) return;
  int s = src[e], d = dst[e];
  float nrm = dinv[s] * dinv[d] * msk;
  atomicAdd(&agg[(size_t)d * kHid + f], hw[(size_t)s * kHid + f] * nrm);
}

// h = relu(agg + hw*dinv^2 + b), in place in agg
__global__ void k_combine_relu(const float* __restrict__ hw, const float* __restrict__ dinv,
                               const float* __restrict__ bias, float* __restrict__ agg,
                               long long total) {
  long long i = (long long)blockIdx.x * blockDim.x + threadIdx.x;
  if (i >= total) return;
  int node = (int)(i >> 7);
  int f = (int)(i & (kHid - 1));
  float di = dinv[node];
  float v = agg[i] + hw[i] * di * di + bias[f];
  agg[i] = fmaxf(v, 0.0f);
}

// ---------------------------------------------------------------- pooling
__global__ void k_dot_wp(const float* __restrict__ h, const float* __restrict__ wp,
                         float* __restrict__ hwp, int n) {
  int i = blockIdx.x * blockDim.x + threadIdx.x;
  if (i >= n) return;
  float s = 0.0f;
#pragma unroll 8
  for (int f = 0; f < kHid; ++f) s += h[(size_t)i * kHid + f] * wp[f];
  hwp[i] = s;
}

__global__ void k_edge_scalar_agg(const int* __restrict__ src, const int* __restrict__ dst,
                                  const float* __restrict__ emask, const float* __restrict__ dinv,
                                  const float* __restrict__ hwp, float* __restrict__ aggp) {
  int e = blockIdx.x * blockDim.x + threadIdx.x;
  if (e >= kE) return;
  float m = emask[e];
  if (m == 0.0f) return;
  int s = src[e], d = dst[e];
  atomicAdd(&aggp[d], hwp[s] * dinv[s] * dinv[d] * m);
}

__global__ void k_score_tanh(const float* __restrict__ aggp, const float* __restrict__ hwp,
                             const float* __restrict__ dinv, const float* __restrict__ bp,
                             float* __restrict__ score, int n) {
  int i = blockIdx.x * blockDim.x + threadIdx.x;
  if (i >= n) return;
  float di = dinv[i];
  score[i] = tanhf(aggp[i] + hwp[i] * di * di + bp[0]);
}

// Per-graph stable top-k via rank counting (matches jax.lax.top_k ordering).
__global__ void k_topk(const float* __restrict__ score, int n_per, int k,
                       int* __restrict__ perm, float* __restrict__ vals) {
  __shared__ float s[512];
  int b = blockIdx.x;
  const float* sc = score + (size_t)b * n_per;
  for (int j = threadIdx.x; j < n_per; j += blockDim.x) s[j] = sc[j];
  __syncthreads();
  for (int j = threadIdx.x; j < n_per; j += blockDim.x) {
    float v = s[j];
    int rank = 0;
    for (int i = 0; i < n_per; ++i) {
      float u = s[i];
      rank += (u > v) || (u == v && i < j);
    }
    if (rank < k) {
      perm[(size_t)b * k + rank] = b * n_per + j;   // old global node id
      vals[(size_t)b * k + rank] = v;
    }
  }
}

__global__ void k_fill_map(int* __restrict__ mapping, int n) {
  int i = blockIdx.x * blockDim.x + threadIdx.x;
  if (i < n) mapping[i] = -1;
}

__global__ void k_set_map(int* __restrict__ mapping, const int* __restrict__ perm, int nk) {
  int i = blockIdx.x * blockDim.x + threadIdx.x;
  if (i < nk) mapping[perm[i]] = i;
}

__global__ void k_gather_scale(const float* __restrict__ h, const int* __restrict__ perm,
                               const float* __restrict__ vals, float* __restrict__ out,
                               long long total) {
  long long i = (long long)blockIdx.x * blockDim.x + threadIdx.x;
  if (i >= total) return;
  int f = (int)(i & (kHid - 1));
  long long node = i >> 7;
  out[i] = h[(size_t)perm[node] * kHid + f] * vals[node];
}

__global__ void k_relabel(int* __restrict__ src, int* __restrict__ dst,
                          float* __restrict__ emask, const int* __restrict__ mapping) {
  int e = blockIdx.x * blockDim.x + threadIdx.x;
  if (e >= kE) return;
  int ns = mapping[src[e]];
  int nd = mapping[dst[e]];
  float m = emask[e] * ((ns >= 0 && nd >= 0) ? 1.0f : 0.0f);
  src[e] = ns > 0 ? ns : 0;
  dst[e] = nd > 0 ? nd : 0;
  emask[e] = m;
}

// z[b, 0:128] += max_i x[b,i,:] ; z[b, 128:256] += mean_i x[b,i,:]
__global__ void k_readout_add(const float* __restrict__ x, int k, float* __restrict__ z) {
  int b = blockIdx.x, f = threadIdx.x;
  const float* xb = x + (size_t)b * k * kHid;
  float mx = -INFINITY, sum = 0.0f;
  for (int i = 0; i < k; ++i) {
    float v = xb[(size_t)i * kHid + f];
    mx = fmaxf(mx, v);
    sum += v;
  }
  z[(size_t)b * 2 * kHid + f] += mx;
  z[(size_t)b * 2 * kHid + kHid + f] += sum / (float)k;
}

// ---------------------------------------------------------------- MLP tail
__global__ void k_bias_relu(float* __restrict__ xb, const float* __restrict__ bias,
                            int total, int ncol) {
  int i = blockIdx.x * blockDim.x + threadIdx.x;
  if (i >= total) return;
  xb[i] = fmaxf(xb[i] + bias[i % ncol], 0.0f);
}

__global__ void k_logsoftmax(const float* __restrict__ z, float* __restrict__ out, int ncol) {
  __shared__ float red[64];
  int b = blockIdx.x, t = threadIdx.x;
  float v = z[(size_t)b * ncol + t];
  red[t] = v;
  __syncthreads();
  for (int s = 32; s > 0; s >>= 1) {
    if (t < s) red[t] = fmaxf(red[t], red[t + s]);
    __syncthreads();
  }
  float mx = red[0];
  __syncthreads();
  red[t] = expf(v - mx);
  __syncthreads();
  for (int s = 32; s > 0; s >>= 1) {
    if (t < s) red[t] += red[t + s];
    __syncthreads();
  }
  float lse = logf(red[0]);
  out[(size_t)b * ncol + t] = v - mx - lse;
}

inline int cdiv(long long n, int bs) { return (int)((n + bs - 1) / bs); }

}  // namespace

extern "C" void kernel_launch(void* const* d_in, const int* in_sizes, int n_in,
                              void* d_out, int out_size, void* d_ws, size_t ws_size,
                              hipStream_t stream) {
  (void)in_sizes; (void)n_in; (void)out_size; (void)ws_size;
  const float* x    = (const float*)d_in[0];
  const int*   src0 = (const int*)d_in[1];
  const int*   dst0 = (const int*)d_in[2];
  const float* W1  = (const float*)d_in[3];
  const float* b1  = (const float*)d_in[4];
  const float* W2  = (const float*)d_in[5];
  const float* b2  = (const float*)d_in[6];
  const float* W3  = (const float*)d_in[7];
  const float* b3  = (const float*)d_in[8];
  const float* Wp  = (const float*)d_in[9];
  const float* bp  = (const float*)d_in[10];
  const float* L1W = (const float*)d_in[11];
  const float* L1b = (const float*)d_in[12];
  const float* L2W = (const float*)d_in[13];
  const float* L2b = (const float*)d_in[14];

  // ----- workspace layout (~53 MB) -----
  size_t off = 0;
  auto alloc = [&](size_t bytes) -> void* {
    void* p = (char*)d_ws + off;
    off += (bytes + 255) & ~(size_t)255;
    return p;
  };
  const int n1 = kB * kN;                                   // 32768
  float* HW    = (float*)alloc((size_t)n1 * kHid * 4);      // XW
  float* AGG   = (float*)alloc((size_t)n1 * kHid * 4);      // agg -> h (in place)
  float* F     = (float*)alloc((size_t)(n1 / 2) * kHid * 4);// pooled features
  _Float16* Ah = (_Float16*)alloc((size_t)(n1 / 2) * kHid * 2); // f16 staged A (max 16384*128)
  _Float16* Wth= (_Float16*)alloc((size_t)kHid * 2 * kHid * 2); // f16 staged W^T (max 128*256)
  int*   csrc  = (int*)alloc((size_t)kE * 4);
  int*   cdst  = (int*)alloc((size_t)kE * 4);
  float* emask = (float*)alloc((size_t)kE * 4);
  float* deg   = (float*)alloc((size_t)n1 * 4);
  float* dinv  = (float*)alloc((size_t)n1 * 4);
  float* hwp   = (float*)alloc((size_t)n1 * 4);
  float* aggp  = (float*)alloc((size_t)n1 * 4);
  float* score = (float*)alloc((size_t)n1 * 4);
  int*   mapping = (int*)alloc((size_t)n1 * 4);
  int*   perm  = (int*)alloc((size_t)(n1 / 2) * 4);
  float* vals  = (float*)alloc((size_t)(n1 / 2) * 4);
  float* zbuf  = (float*)alloc((size_t)kB * 2 * kHid * 4);
  float* z1    = (float*)alloc((size_t)kB * kHid * 4);
  float* z2    = (float*)alloc((size_t)kB * (kHid / 2) * 4);

  const int BS = 256;

  // C[M,N] = A[M,K] * W[K,N] : stage to f16 (K padded to 32) then WMMA
  auto gemm = [&](const float* A, const float* Wm, float* Cm, int M, int K, int N) {
    const int Kp = (K + 31) & ~31;
    const long long ta = (long long)M * Kp;
    const long long tw = (long long)N * Kp;
    k_pack_a<<<cdiv(ta, BS), BS, 0, stream>>>(A, Ah, ta, K, Kp);
    k_pack_wt<<<cdiv(tw, BS), BS, 0, stream>>>(Wm, Wth, tw, K, N, Kp);
    dim3 gr(M / 16, (N + 63) / 64);
    if (Kp == 32) {
      k_gemm_wmma_h<32><<<gr, 128, 0, stream>>>(Ah, Wth, Cm, N);
    } else if (Kp == 128) {
      k_gemm_wmma_h<128><<<gr, 128, 0, stream>>>(Ah, Wth, Cm, N);
    } else {  // 256
      k_gemm_wmma_h<256><<<gr, 128, 0, stream>>>(Ah, Wth, Cm, N);
    }
  };

  k_init_edges<<<cdiv(kE, BS), BS, 0, stream>>>(src0, dst0, csrc, cdst, emask);
  k_zero_f32<<<cdiv(kB * 2 * kHid, BS), BS, 0, stream>>>(zbuf, kB * 2 * kHid);

  auto layer = [&](const float* feats, int K, const float* Wl, const float* bl, int n_per) {
    const int n = kB * n_per;
    const int kkeep = n_per / 2;
    // ---- GCN(feats) -> AGG (relu'd) ----
    k_zero_f32<<<cdiv(n, BS), BS, 0, stream>>>(deg, n);
    k_deg_accum<<<cdiv(kE, BS), BS, 0, stream>>>(cdst, emask, deg);
    k_dinv<<<cdiv(n, BS), BS, 0, stream>>>(deg, dinv, n);
    gemm(feats, Wl, HW, n, K, kHid);
    const long long nf = (long long)n * kHid;
    k_zero_f32<<<cdiv(nf, BS), BS, 0, stream>>>(AGG, nf);
    const long long ef = (long long)kE * kHid;
    k_edge_feat_agg<<<cdiv(ef, BS), BS, 0, stream>>>(csrc, cdst, emask, dinv, HW, AGG, ef);
    k_combine_relu<<<cdiv(nf, BS), BS, 0, stream>>>(HW, dinv, bl, AGG, nf);
    // ---- SAGPool: score = tanh(GCN(h, Wp, bp)) on same graph ----
    k_dot_wp<<<cdiv(n, BS), BS, 0, stream>>>(AGG, Wp, hwp, n);
    k_zero_f32<<<cdiv(n, BS), BS, 0, stream>>>(aggp, n);
    k_edge_scalar_agg<<<cdiv(kE, BS), BS, 0, stream>>>(csrc, cdst, emask, dinv, hwp, aggp);
    k_score_tanh<<<cdiv(n, BS), BS, 0, stream>>>(aggp, hwp, dinv, bp, score, n);
    k_topk<<<kB, 256, 0, stream>>>(score, n_per, kkeep, perm, vals);
    k_fill_map<<<cdiv(n, BS), BS, 0, stream>>>(mapping, n);
    k_set_map<<<cdiv(kB * kkeep, BS), BS, 0, stream>>>(mapping, perm, kB * kkeep);
    const long long pf = (long long)kB * kkeep * kHid;
    k_gather_scale<<<cdiv(pf, BS), BS, 0, stream>>>(AGG, perm, vals, F, pf);
    k_relabel<<<cdiv(kE, BS), BS, 0, stream>>>(csrc, cdst, emask, mapping);
    // ---- readout (max || mean) accumulated into zbuf ----
    k_readout_add<<<kB, kHid, 0, stream>>>(F, kkeep, zbuf);
  };

  layer(x, 10,   W1, b1, kN);       // 512 nodes/graph -> 256
  layer(F, kHid, W2, b2, kN / 2);   // 256 -> 128
  layer(F, kHid, W3, b3, kN / 4);   // 128 -> 64

  // ---- MLP head: relu(z@L1W+b) -> relu(@L2W+b) -> log_softmax ----
  gemm(zbuf, L1W, z1, kB, 2 * kHid, kHid);
  k_bias_relu<<<cdiv(kB * kHid, BS), BS, 0, stream>>>(z1, L1b, kB * kHid, kHid);
  gemm(z1, L2W, z2, kB, kHid, kHid / 2);
  k_bias_relu<<<cdiv(kB * (kHid / 2), BS), BS, 0, stream>>>(z2, L2b, kB * (kHid / 2), kHid / 2);
  k_logsoftmax<<<kB, 64, 0, stream>>>(z2, (float*)d_out, kHid / 2);
}
